// Net_86122684219654
// MI455X (gfx1250) — compile-verified, matrix operand order checked by reading, not run on hardware
//
#include <hip/hip_runtime.h>
#include <hip/hip_bf16.h>
#include <math.h>
#include <stdint.h>

// ============================================================================
// MI455X (gfx1250) implementation of the PointCNN-ish reference network.
// Dense contractions -> v_wmma_f32_16x16x32_f16 (f16 mul, f32 accumulate),
// A tiles staged global->LDS with global_load_async_to_lds_b128 (ASYNCcnt),
// B pre-packed to zero-padded f16 [n][k].  Irregular parts are VALU kernels.
//
// Parameter flattening assumption: top-level setup_inputs() insertion order
// (pos, batch, params), with `params` flattened as a JAX pytree (dict keys in
// sorted ASCII order, lists in index order).
// ============================================================================

typedef __attribute__((ext_vector_type(16))) _Float16 v16h;
typedef __attribute__((ext_vector_type(8)))  float    v8f;
typedef __attribute__((ext_vector_type(4)))  float    f32x4;
typedef __attribute__((ext_vector_type(8)))  _Float16 f16x8;

#define B_SZ 32

__device__ __forceinline__ float eluf(float x){ return x > 0.0f ? x : expm1f(x); }
__device__ __forceinline__ float sigf(float x){ return 1.0f/(1.0f+expf(-x)); }

// CDNA5 async global->LDS copy (16B per lane), tracked by ASYNCcnt.
__device__ __forceinline__ void async_copy_b128(uint32_t lds_off, const float* gaddr)
{
    asm volatile("global_load_async_to_lds_b128 %0, %1, off"
                 :: "v"(lds_off), "v"(gaddr) : "memory");
}
__device__ __forceinline__ void wait_async0()
{
    asm volatile("s_wait_asynccnt 0" ::: "memory");
}

// ---------------------------------------------------------------------------
// Pack B (KxN f32, row-major) -> Bt (N x Kp f16), k zero-padded to Kp (%32==0).
// Padding guarantees exact-zero products for the K tail in the WMMA loop.
// ---------------------------------------------------------------------------
__global__ void packB_kernel(const float* __restrict__ B, _Float16* __restrict__ Bt,
                             int K, int N, int Kp)
{
    const int id = blockIdx.x * blockDim.x + threadIdx.x;
    if (id >= N * Kp) return;
    const int k = id % Kp, n = id / Kp;
    Bt[id] = (k < K) ? (_Float16)B[(size_t)k * N + n] : (_Float16)0.0f;
}

// ---------------------------------------------------------------------------
// WMMA GEMM:  C = act(A(MxK,f32) @ B + bias) * gamma + beta
// Block = 4 waves sharing one 16-row M tile staged in LDS (async-to-LDS),
// each wave owns one 16-col N tile.  B read from packed f16 Bt (N x Kp).
// Fragment layouts per CDNA5 ISA 7.12.2.  ACT: 0=none, 1=relu, 2=elu.
// ---------------------------------------------------------------------------
template<int ACT, bool HAS_SCALE>
__global__ __launch_bounds__(128)
void wmma_gemm_kernel(const float* __restrict__ A, const _Float16* __restrict__ Bt,
                      const float* __restrict__ bias,
                      const float* __restrict__ gamma, const float* __restrict__ beta,
                      float* __restrict__ Cmat, int M, int K, int Kp, int N)
{
    __shared__ __align__(16) float tileA[16 * 32];      // one 16x32 f32 panel

    const int lane = threadIdx.x;                       // 0..31
    const int wave = threadIdx.y;                       // 0..3
    const int tid  = wave * 32 + lane;                  // 0..127
    const int tm   = blockIdx.x;
    const int tn   = blockIdx.y * 4 + wave;
    const int tilesN = (N + 15) >> 4;
    const bool tnValid = (tn < tilesN);

    const int half = lane >> 4;                         // 0/1 lane halves
    const int l15  = lane & 15;
    const int cA   = half * 8;                          // per-half k offset

    const int colB = min(tn * 16 + l15, N - 1);         // clamped (masked at store)
    const _Float16* BtRow = Bt + (size_t)colB * Kp;

    // cooperative A staging: thread -> (row 0..15, 16B chunk 0..7)
    const int arow = tid >> 3;
    const int aq   = (tid & 7) * 4;
    const int grow = min(tm * 16 + arow, M - 1);        // clamp M edge (finite data)
    const float* gRow = A + (size_t)grow * K;
    float* ldsDst = &tileA[arow * 32 + aq];
    const uint32_t ldsOff = (uint32_t)(uintptr_t)ldsDst;

    v8f acc = {};
    for (int k0 = 0; k0 < Kp; k0 += 32) {
        const int gk = k0 + aq;
        if (gk + 4 <= K) {
            async_copy_b128(ldsOff, gRow + gk);         // hot path: no guards
        } else {
#pragma unroll
            for (int j = 0; j < 4; ++j)                 // K tail: exact zeros
                ldsDst[j] = (gk + j < K) ? gRow[gk + j] : 0.0f;
        }
        wait_async0();
        __syncthreads();

        if (tnValid) {
            const f32x4* Ap0 = (const f32x4*)&tileA[l15 * 32 + cA];
            const f32x4* Ap1 = (const f32x4*)&tileA[l15 * 32 + 16 + cA];
            const f32x4 a0 = Ap0[0], a1 = Ap0[1];       // k = cA .. cA+7
            const f32x4 a2 = Ap1[0], a3 = Ap1[1];       // k = 16+cA .. 16+cA+7
            const f16x8 blo = *(const f16x8*)(BtRow + k0 + cA);
            const f16x8 bhi = *(const f16x8*)(BtRow + k0 + 16 + cA);
            v16h a, b;
#pragma unroll
            for (int j = 0; j < 4; ++j) {
                a[j]      = (_Float16)a0[j];
                a[4 + j]  = (_Float16)a1[j];
                a[8 + j]  = (_Float16)a2[j];
                a[12 + j] = (_Float16)a3[j];
            }
#pragma unroll
            for (int j = 0; j < 8; ++j) { b[j] = blo[j]; b[8 + j] = bhi[j]; }
            acc = __builtin_amdgcn_wmma_f32_16x16x32_f16(
                      false, a, false, b, (short)0, acc, false, false);
        }
        __syncthreads();
    }

    if (!tnValid) return;
#pragma unroll
    for (int r = 0; r < 8; ++r) {
        const int m = tm * 16 + r + (half ? 8 : 0);
        const int n = tn * 16 + l15;
        if (m < M && n < N) {
            float v = acc[r];
            if (bias)     v += bias[n];
            if (ACT == 1) v = fmaxf(v, 0.0f);
            if (ACT == 2) v = eluf(v);
            if (HAS_SCALE) v = v * gamma[n] + beta[n];
            Cmat[(size_t)m * N + n] = v;
        }
    }
}

static void gemm(hipStream_t s, const float* A, const float* Bm, const float* bias,
                 const float* g, const float* be, float* C,
                 int M, int K, int N, int act, _Float16* Bt)
{
    const int Kp = (K + 31) & ~31;
    const int pthr = N * Kp;
    packB_kernel<<<(pthr + 255)/256, 256, 0, s>>>(Bm, Bt, K, N, Kp);
    dim3 blk(32, 4, 1);
    dim3 grd((unsigned)((M + 15) / 16), (unsigned)((((N + 15) / 16) + 3) / 4), 1);
    if (g) {
        if (act == 2)      wmma_gemm_kernel<2,true ><<<grd,blk,0,s>>>(A,Bt,bias,g,be,C,M,K,Kp,N);
        else if (act == 1) wmma_gemm_kernel<1,true ><<<grd,blk,0,s>>>(A,Bt,bias,g,be,C,M,K,Kp,N);
        else               wmma_gemm_kernel<0,true ><<<grd,blk,0,s>>>(A,Bt,bias,g,be,C,M,K,Kp,N);
    } else {
        if (act == 2)      wmma_gemm_kernel<2,false><<<grd,blk,0,s>>>(A,Bt,bias,g,be,C,M,K,Kp,N);
        else if (act == 1) wmma_gemm_kernel<1,false><<<grd,blk,0,s>>>(A,Bt,bias,g,be,C,M,K,Kp,N);
        else               wmma_gemm_kernel<0,false><<<grd,blk,0,s>>>(A,Bt,bias,g,be,C,M,K,Kp,N);
    }
}

// ---------------------------------------------------------------------------
// KNN: per query point, keep K*dil nearest (ascending distance, lower index
// wins ties, self included) then take every dil-th -> K neighbor indices.
// ---------------------------------------------------------------------------
__global__ void knn_kernel(const float* __restrict__ pos, int* __restrict__ nbr,
                           int N, int K, int dil)
{
    const int id = blockIdx.x * blockDim.x + threadIdx.x;
    if (id >= B_SZ * N) return;
    const int b = id / N, n = id % N;
    const float* P = pos + (size_t)b * N * 3;
    const float qx = P[n*3+0], qy = P[n*3+1], qz = P[n*3+2];
    const int KD = K * dil;                       // <= 32
    float bd[32]; int bi[32];
    for (int i = 0; i < KD; ++i) { bd[i] = 3.402823466e38f; bi[i] = 0; }
    for (int m = 0; m < N; ++m) {
        const float dx = P[m*3+0]-qx, dy = P[m*3+1]-qy, dz = P[m*3+2]-qz;
        const float d = dx*dx + dy*dy + dz*dz;
        if (d < bd[KD-1]) {
            int j = KD - 1;
            while (j > 0 && bd[j-1] > d) { bd[j] = bd[j-1]; bi[j] = bi[j-1]; --j; }
            bd[j] = d; bi[j] = m;
        }
    }
    int* out = nbr + (size_t)id * K;
    for (int k = 0; k < K; ++k) out[k] = bi[k * dil];
}

// ---------------------------------------------------------------------------
// Stage 1: rel = pos[nbr]-pos ; h1 = elu(rel@W1+b1)*g1+be1   (K-dim is 3)
// ---------------------------------------------------------------------------
__global__ void stage1_kernel(const float* __restrict__ pos, const int* __restrict__ nbr,
                              const float* __restrict__ W1, const float* __restrict__ b1,
                              const float* __restrict__ g1, const float* __restrict__ be1,
                              float* __restrict__ rel, float* __restrict__ h1,
                              int N, int K, int Cd)
{
    const int id = blockIdx.x * blockDim.x + threadIdx.x;
    if (id >= B_SZ * N * K) return;
    const int k  = id % K;
    const int bn = id / K;
    const int b  = bn / N, n = bn % N;
    const float* P = pos + (size_t)b * N * 3;
    const int j = nbr[(size_t)bn * K + k];
    const float rx = P[j*3+0] - P[n*3+0];
    const float ry = P[j*3+1] - P[n*3+1];
    const float rz = P[j*3+2] - P[n*3+2];
    float* rp = rel + (size_t)id * 3;
    rp[0] = rx; rp[1] = ry; rp[2] = rz;
    float* hp = h1 + (size_t)id * Cd;
    for (int c = 0; c < Cd; ++c) {
        float h = rx * W1[c] + ry * W1[Cd + c] + rz * W1[2*Cd + c] + b1[c];
        h = eluf(h);
        hp[c] = h * g1[c] + be1[c];
    }
}

// ---------------------------------------------------------------------------
// T-matrix chain: u = (elu(sum_l t[g,l]*c1W[g,j,l]+c1b))*g2+be2
//                 T = (sum_l u[g,l]*c2W[g,j,l]+c2b)*g3+be3
// ---------------------------------------------------------------------------
__global__ void tmat_kernel(const float* __restrict__ t1,
                            const float* __restrict__ c1W, const float* __restrict__ c1b,
                            const float* __restrict__ g2,  const float* __restrict__ be2,
                            const float* __restrict__ c2W, const float* __restrict__ c2b,
                            const float* __restrict__ g3,  const float* __restrict__ be3,
                            float* __restrict__ Tm, int BN, int K)
{
    const int id = blockIdx.x * blockDim.x + threadIdx.x;
    if (id >= BN * K) return;
    const int g = id % K;
    const size_t pnt = (size_t)(id / K);
    const float* trow = t1 + (pnt * K + g) * K;
    float u[16];
    for (int j = 0; j < K; ++j) {
        float s = c1b[g*K + j];
        const float* wp = c1W + ((size_t)(g*K + j)) * K;
        for (int l = 0; l < K; ++l) s += trow[l] * wp[l];
        s = eluf(s);
        u[j] = s * g2[g*K + j] + be2[g*K + j];
    }
    float* orow = Tm + (pnt * K + g) * K;
    for (int j = 0; j < K; ++j) {
        float s = c2b[g*K + j];
        const float* wp = c2W + ((size_t)(g*K + j)) * K;
        for (int l = 0; l < K; ++l) s += u[l] * wp[l];
        orow[j] = s * g3[g*K + j] + be3[g*K + j];
    }
}

// ---------------------------------------------------------------------------
// Fused: build x_star row c on the fly (h2 part + gathered x part),
// xt = x_star @ T (KxK), then y[c*dm+j] = sum_l xt[l]*convW[c,j,l] + convb.
// ---------------------------------------------------------------------------
__global__ void xtconv_kernel(const float* __restrict__ h2, const float* __restrict__ xin,
                              const int* __restrict__ nbr, const float* __restrict__ Tm,
                              const float* __restrict__ convW, const float* __restrict__ convb,
                              float* __restrict__ y,
                              int BN, int Nl, int K, int Cd, int Cin, int dm)
{
    const int C = Cd + Cin;
    const int id = blockIdx.x * blockDim.x + threadIdx.x;
    if (id >= BN * C) return;
    const int c = id % C;
    const size_t pnt = (size_t)(id / C);
    const int b = (int)(pnt / Nl);
    const int* nb = nbr + pnt * K;
    float sc[16];
    for (int k = 0; k < K; ++k) {
        sc[k] = (c < Cd) ? h2[(pnt * K + k) * (size_t)Cd + c]
                         : xin[((size_t)b * Nl + nb[k]) * Cin + (c - Cd)];
    }
    float xt[16];
    const float* Tp = Tm + pnt * K * K;
    for (int l = 0; l < K; ++l) {
        float s = 0.0f;
        for (int k = 0; k < K; ++k) s += sc[k] * Tp[k*K + l];
        xt[l] = s;
    }
    for (int j = 0; j < dm; ++j) {
        float s = convb[c*dm + j];
        const float* wp = convW + ((size_t)(c*dm + j)) * K;
        for (int l = 0; l < K; ++l) s += xt[l] * wp[l];
        y[pnt * (size_t)(C*dm) + c*dm + j] = s;
    }
}

// ---------------------------------------------------------------------------
// Farthest point sampling; one block per batch, matches jax scan semantics.
// ---------------------------------------------------------------------------
__global__ void fps_kernel(const float* __restrict__ pos, int* __restrict__ idx,
                           int N, int M)
{
    __shared__ float mind[1024];
    __shared__ float rd[256];
    __shared__ int   ri[256];
    __shared__ int   s_last;
    const int b = blockIdx.x, t = threadIdx.x;
    const float* P = pos + (size_t)b * N * 3;
    for (int i = t; i < N; i += 256) mind[i] = 3.402823466e38f;
    if (t == 0) { idx[(size_t)b * M] = 0; s_last = 0; }
    __syncthreads();
    int last = 0;
    for (int m = 1; m < M; ++m) {
        const float lx = P[last*3+0], ly = P[last*3+1], lz = P[last*3+2];
        float bdv = -1.0f; int biv = N;
        for (int i = t; i < N; i += 256) {
            const float dx = P[i*3+0]-lx, dy = P[i*3+1]-ly, dz = P[i*3+2]-lz;
            const float md = fminf(mind[i], dx*dx + dy*dy + dz*dz);
            mind[i] = md;
            if (md > bdv || (md == bdv && i < biv)) { bdv = md; biv = i; }
        }
        rd[t] = bdv; ri[t] = biv;
        __syncthreads();
        for (int s = 128; s > 0; s >>= 1) {
            if (t < s) {
                if (rd[t+s] > rd[t] || (rd[t+s] == rd[t] && ri[t+s] < ri[t])) {
                    rd[t] = rd[t+s]; ri[t] = ri[t+s];
                }
            }
            __syncthreads();
        }
        if (t == 0) { idx[(size_t)b * M + m] = ri[0]; s_last = ri[0]; }
        __syncthreads();
        last = s_last;
    }
}

__global__ void gather_kernel(const float* __restrict__ src, const int* __restrict__ idx,
                              float* __restrict__ dst, int Nsrc, int Msel, int Cch)
{
    const int id = blockIdx.x * blockDim.x + threadIdx.x;
    if (id >= B_SZ * Msel * Cch) return;
    const int c  = id % Cch;
    const int bm = id / Cch;
    const int b  = bm / Msel, m = bm % Msel;
    dst[id] = src[((size_t)b * Nsrc + idx[(size_t)b * Msel + m]) * Cch + c];
}

__global__ void mean_kernel(const float* __restrict__ x, float* __restrict__ out,
                            int Nl, int Cc)
{
    const int id = blockIdx.x * blockDim.x + threadIdx.x;
    if (id >= B_SZ * Cc) return;
    const int c = id % Cc, b = id / Cc;
    float s = 0.0f;
    for (int n = 0; n < Nl; ++n) s += x[((size_t)b * Nl + n) * Cc + c];
    out[id] = s / (float)Nl;
}

// ---------------------------------------------------------------------------
// Two stacked bidirectional LSTMs, H=1, T=192.  One block, 64 threads.
// ---------------------------------------------------------------------------
__global__ void lstm_head_kernel(const float* __restrict__ xin,
    const float* w0bw_hh, const float* w0bw_ih, const float* b0bw_hh, const float* b0bw_ih,
    const float* w0fw_hh, const float* w0fw_ih, const float* b0fw_hh, const float* b0fw_ih,
    const float* w1bw_hh, const float* w1bw_ih, const float* b1bw_hh, const float* b1bw_ih,
    const float* w1fw_hh, const float* w1fw_ih, const float* b1fw_hh, const float* b1fw_ih,
    float* __restrict__ out)
{
    __shared__ float hbuf[B_SZ][192][2];          // 48 KB LDS
    const int T = 192;
    const int t = threadIdx.x;                    // 0..63
    const int b = t >> 1, dir = t & 1;            // dir 0 = fw, 1 = bw
    {   // layer 1, input feature = 1
        const float* Whh = dir ? w0bw_hh : w0fw_hh;
        const float* Wih = dir ? w0bw_ih : w0fw_ih;
        const float* bhh = dir ? b0bw_hh : b0fw_hh;
        const float* bih = dir ? b0bw_ih : b0fw_ih;
        float h = 0.0f, c = 0.0f;
        for (int s = 0; s < T; ++s) {
            const int tt = dir ? (T - 1 - s) : s;
            const float xv = xin[b*T + tt];
            const float zi = Wih[0]*xv + Whh[0]*h + bih[0] + bhh[0];
            const float zf = Wih[1]*xv + Whh[1]*h + bih[1] + bhh[1];
            const float zg = Wih[2]*xv + Whh[2]*h + bih[2] + bhh[2];
            const float zo = Wih[3]*xv + Whh[3]*h + bih[3] + bhh[3];
            c = sigf(zf)*c + sigf(zi)*tanhf(zg);
            h = sigf(zo)*tanhf(c);
            hbuf[b][tt][dir] = h;
        }
    }
    __syncthreads();
    if (dir == 0) {                               // layer 2 forward, feat = 2
        float h = 0.0f, c = 0.0f;
        for (int tt = 0; tt < T; ++tt) {
            const float x0 = hbuf[b][tt][0], x1 = hbuf[b][tt][1];
            const float zi = w1fw_ih[0]*x0 + w1fw_ih[1]*x1 + w1fw_hh[0]*h + b1fw_ih[0] + b1fw_hh[0];
            const float zf = w1fw_ih[2]*x0 + w1fw_ih[3]*x1 + w1fw_hh[1]*h + b1fw_ih[1] + b1fw_hh[1];
            const float zg = w1fw_ih[4]*x0 + w1fw_ih[5]*x1 + w1fw_hh[2]*h + b1fw_ih[2] + b1fw_hh[2];
            const float zo = w1fw_ih[6]*x0 + w1fw_ih[7]*x1 + w1fw_hh[3]*h + b1fw_ih[3] + b1fw_hh[3];
            c = sigf(zf)*c + sigf(zi)*tanhf(zg);
            h = sigf(zo)*tanhf(c);
            out[b*T + tt] = 0.5f * h;             // mean over last axis, part 1
        }
    }
    __threadfence_block();
    __syncthreads();
    if (dir == 1) {                               // layer 2 backward
        float h = 0.0f, c = 0.0f;
        for (int s = 0; s < T; ++s) {
            const int tt = T - 1 - s;
            const float x0 = hbuf[b][tt][0], x1 = hbuf[b][tt][1];
            const float zi = w1bw_ih[0]*x0 + w1bw_ih[1]*x1 + w1bw_hh[0]*h + b1bw_ih[0] + b1bw_hh[0];
            const float zf = w1bw_ih[2]*x0 + w1bw_ih[3]*x1 + w1bw_hh[1]*h + b1bw_ih[1] + b1bw_hh[1];
            const float zg = w1bw_ih[4]*x0 + w1bw_ih[5]*x1 + w1bw_hh[2]*h + b1bw_ih[2] + b1bw_hh[2];
            const float zo = w1bw_ih[6]*x0 + w1bw_ih[7]*x1 + w1bw_hh[3]*h + b1bw_ih[3] + b1bw_hh[3];
            c = sigf(zf)*c + sigf(zi)*tanhf(zg);
            h = sigf(zo)*tanhf(c);
            out[b*T + tt] += 0.5f * h;            // mean over last axis, part 2
        }
    }
}

__global__ void logsoftmax_kernel(const float* __restrict__ logits, float* __restrict__ out)
{
    const int b = threadIdx.x;
    if (b >= B_SZ) return;
    const float* r = logits + b * 40;
    float mx = r[0];
    for (int i = 1; i < 40; ++i) mx = fmaxf(mx, r[i]);
    float se = 0.0f;
    for (int i = 0; i < 40; ++i) se += expf(r[i] - mx);
    const float lse = mx + logf(se);
    for (int i = 0; i < 40; ++i) out[b*40 + i] = r[i] - lse;
}

// ============================================================================
// Host-side orchestration
// ============================================================================
struct XCfg { int Cin, Cout, K, Cd, dil, N; };
static const XCfg g_cfg[5] = {
    {  0,  48,  8,  32, 1, 1024},
    { 48,  96, 12,  64, 2,  384},
    { 96, 192, 16, 128, 2,  129},
    {192, 384, 16, 256, 2,  129},
    {384, 192, 16, 256, 2,  129},
};

// sorted-key leaf order inside each conv param dict
enum { P_c1_W = 0, P_c1_b, P_c2_W, P_c2_b, P_conv_W, P_conv_b, P_lin_W, P_lin_b,
       P_m1_W1, P_m1_W2, P_m1_b1, P_m1_b2, P_m1_be1, P_m1_be2, P_m1_g1, P_m1_g2,
       P_m2_W, P_m2_b, P_m2_be1, P_m2_be2, P_m2_be3, P_m2_g1, P_m2_g2, P_m2_g3 };

extern "C" void kernel_launch(void* const* d_in, const int* in_sizes, int n_in,
                              void* d_out, int out_size, void* d_ws, size_t ws_size,
                              hipStream_t stream)
{
    (void)in_sizes; (void)n_in; (void)out_size; (void)ws_size;
    const float* pos0 = (const float*)d_in[0];

    // ---- workspace arena ------------------------------------------------
    char* wsb = (char*)d_ws;
    size_t off = 0;
    auto alloc = [&](size_t nbytes) -> void* {
        void* p = (void*)(wsb + off);
        off += (nbytes + 255) & ~(size_t)255;
        return p;
    };
    float*    h1    = (float*)   alloc(16908288ull * 4);   // B*129*16*256 max
    float*    h2    = (float*)   alloc(16908288ull * 4);
    float*    rel   = (float*)   alloc(  786432ull * 4);   // B*1024*8*3 max
    float*    t1    = (float*)   alloc( 2097152ull * 4);   // B*1024*64 max
    float*    Tm    = (float*)   alloc( 2097152ull * 4);
    float*    ybuf  = (float*)   alloc( 2641920ull * 4);   // B*129*640 max
    float*    xbufA = (float*)   alloc( 1572864ull * 4);   // B*1024*48 max
    float*    xbufB = (float*)   alloc( 1572864ull * 4);
    _Float16* Btp   = (_Float16*)alloc(  262144ull * 2);   // packed f16 weights
    int*      nbr   = (int*)     alloc(  262144ull * 4);   // B*1024*8 max
    int*      idx1  = (int*)     alloc((size_t)B_SZ * 384 * 4);
    int*      idx2  = (int*)     alloc((size_t)B_SZ * 129 * 4);
    float*    posB  = (float*)   alloc((size_t)B_SZ * 384 * 3 * 4);
    float*    posC  = (float*)   alloc((size_t)B_SZ * 129 * 3 * 4);
    float*    xm    = (float*)   alloc((size_t)B_SZ * 192 * 4);
    float*    xl    = (float*)   alloc((size_t)B_SZ * 192 * 4);
    float*    a1    = (float*)   alloc((size_t)B_SZ * 256 * 4);
    float*    a2    = (float*)   alloc((size_t)B_SZ * 128 * 4);
    float*    lg    = (float*)   alloc((size_t)B_SZ *  40 * 4);

    float* outs[5] = { xbufA, xbufA, xbufA, xbufB, xbufA };
    const float* poscur = pos0;
    const float* xcur   = nullptr;

    for (int l = 0; l < 5; ++l) {
        const XCfg& cf = g_cfg[l];
        const int Nl = cf.N, K = cf.K, Cd = cf.Cd, Cin = cf.Cin;
        const int C  = Cin + Cd;
        const int dm = (cf.Cout + C - 1) / C;
        auto CP = [&](int o) -> const float* { return (const float*)d_in[2 + l*24 + o]; };

        int thr = B_SZ * Nl;
        knn_kernel<<<(thr + 255)/256, 256, 0, stream>>>(poscur, nbr, Nl, K, cf.dil);

        thr = B_SZ * Nl * K;
        stage1_kernel<<<(thr + 255)/256, 256, 0, stream>>>(
            poscur, nbr, CP(P_m1_W1), CP(P_m1_b1), CP(P_m1_g1), CP(P_m1_be1),
            rel, h1, Nl, K, Cd);

        // h2 = elu(h1 @ m1_W2 + b2) * g2 + be2         (WMMA)
        gemm(stream, h1, CP(P_m1_W2), CP(P_m1_b2), CP(P_m1_g2), CP(P_m1_be2),
             h2, B_SZ * Nl * K, Cd, Cd, 2, Btp);

        // t1 = elu(rel_flat @ m2_W + b) * g1 + be1     (WMMA)
        gemm(stream, rel, CP(P_m2_W), CP(P_m2_b), CP(P_m2_g1), CP(P_m2_be1),
             t1, B_SZ * Nl, 3 * K, K * K, 2, Btp);

        thr = B_SZ * Nl * K;
        tmat_kernel<<<(thr + 255)/256, 256, 0, stream>>>(
            t1, CP(P_c1_W), CP(P_c1_b), CP(P_m2_g2), CP(P_m2_be2),
            CP(P_c2_W), CP(P_c2_b), CP(P_m2_g3), CP(P_m2_be3),
            Tm, B_SZ * Nl, K);

        thr = B_SZ * Nl * C;
        xtconv_kernel<<<(thr + 255)/256, 256, 0, stream>>>(
            h2, xcur, nbr, Tm, CP(P_conv_W), CP(P_conv_b),
            ybuf, B_SZ * Nl, Nl, K, Cd, Cin, dm);

        // out = relu(y @ lin_W + lin_b)                (WMMA)
        gemm(stream, ybuf, CP(P_lin_W), CP(P_lin_b), nullptr, nullptr,
             outs[l], B_SZ * Nl, C * dm, cf.Cout, 1, Btp);

        if (l == 0) {
            fps_kernel<<<B_SZ, 256, 0, stream>>>(pos0, idx1, 1024, 384);
            int g1 = B_SZ * 384 * 3;
            gather_kernel<<<(g1 + 255)/256, 256, 0, stream>>>(pos0, idx1, posB, 1024, 384, 3);
            int g2 = B_SZ * 384 * 48;
            gather_kernel<<<(g2 + 255)/256, 256, 0, stream>>>(xbufA, idx1, xbufB, 1024, 384, 48);
            poscur = posB; xcur = xbufB;
        } else if (l == 1) {
            fps_kernel<<<B_SZ, 256, 0, stream>>>(posB, idx2, 384, 129);
            int g1 = B_SZ * 129 * 3;
            gather_kernel<<<(g1 + 255)/256, 256, 0, stream>>>(posB, idx2, posC, 384, 129, 3);
            int g2 = B_SZ * 129 * 96;
            gather_kernel<<<(g2 + 255)/256, 256, 0, stream>>>(xbufA, idx2, xbufB, 384, 129, 96);
            poscur = posC; xcur = xbufB;
        } else {
            xcur = outs[l];
        }
    }

    // mean over points -> (B,192)
    {
        int thr = B_SZ * 192;
        mean_kernel<<<(thr + 255)/256, 256, 0, stream>>>(xbufA, xm, 129, 192);
    }

    // LSTM stack: params at indices 128.. : l0{bw,fw}, l1{bw,fw}, each {Whh,Wih,bhh,bih}
    {
        const float* L[16];
        for (int i = 0; i < 16; ++i) L[i] = (const float*)d_in[128 + i];
        lstm_head_kernel<<<1, 64, 0, stream>>>(xm,
            L[0], L[1], L[2], L[3],      // l0 bw
            L[4], L[5], L[6], L[7],      // l0 fw
            L[8], L[9], L[10], L[11],    // l1 bw
            L[12], L[13], L[14], L[15],  // l1 fw
            xl);
    }

    // MLP head (WMMA) + log-softmax
    const float* lin1_W = (const float*)d_in[122];
    const float* lin1_b = (const float*)d_in[123];
    const float* lin2_W = (const float*)d_in[124];
    const float* lin2_b = (const float*)d_in[125];
    const float* lin3_W = (const float*)d_in[126];
    const float* lin3_b = (const float*)d_in[127];
    gemm(stream, xl, lin1_W, lin1_b, nullptr, nullptr, a1, B_SZ, 192, 256, 1, Btp);
    gemm(stream, a1, lin2_W, lin2_b, nullptr, nullptr, a2, B_SZ, 256, 128, 1, Btp);
    gemm(stream, a2, lin3_W, lin3_b, nullptr, nullptr, lg, B_SZ, 128,  40, 0, Btp);
    logsoftmax_kernel<<<1, 32, 0, stream>>>(lg, (float*)d_out);
}
